// CaptionDecoderLSTM_51153060495386
// MI455X (gfx1250) — compile-verified, matrix operand order checked by reading, not run on hardware
//
#include <hip/hip_runtime.h>
#include <hip/hip_bf16.h>
#include <stdint.h>

// ---------------------------------------------------------------------------
// CaptionDecoderLSTM for MI455X (gfx1250, wave32, WMMA bf16->f32)
// All dense projections run through v_wmma_f32_16x16x32_bf16.
// Each wave computes a 16Mx32N tile (2 accumulators) so one A-fragment
// conversion (f32->bf16 cvt_pk) feeds two WMMAs.
// ---------------------------------------------------------------------------

constexpr int kB = 64;     // batch
constexpr int kR = 196;    // regions
constexpr int kF = 2048;   // feature dim
constexpr int kE = 512;    // embed dim
constexpr int kH = 512;    // hidden dim
constexpr int kA = 512;    // attn dim
constexpr int kV = 12000;  // vocab
constexpr int kL = 24;     // seq len
constexpr int kX = kE + kF + kH;  // 3072 : [emb | context | h] concat K for gates GEMM

typedef __attribute__((ext_vector_type(16))) __bf16 v16bf;
typedef __attribute__((ext_vector_type(8)))  float  v8f;

union Frag16 { v16bf v; uint32_t u[8]; };

static __device__ __forceinline__ uint32_t pk2bf(float lo, float hi) {
    union { __bf16 h[2]; uint32_t u; } r;
    r.h[0] = (__bf16)lo;
    r.h[1] = (__bf16)hi;
    return r.u;
}

static __device__ __forceinline__ float sigmoidf_(float x) {
    return 1.0f / (1.0f + __expf(-x));
}

// ---------------------------------------------------------------------------
// Generic GEMM:  C[M,N] = A[M,K](f32, cvt->bf16) * W[N,K](bf16)^T + bias[N]
// One wave per 16x32 C tile (two 16x16 WMMA tiles sharing the A fragment),
// 4 waves/block covering 16M x 128N.  K % 32 == 0, N % 32 == 0 assumed
// (all call sites satisfy this).  Fragment layout per CDNA5 ISA 7.12.2:
//   lane half hk = lane>>4 ; per-lane K chunk base = k0 + 8*hk
//   elems 0..7  -> K = base..base+7 ; elems 8..15 -> K = base+16..base+23
// ---------------------------------------------------------------------------
__global__ __launch_bounds__(128) void gemm_bf16_wmma(
    const float* __restrict__ A, int lda,
    const __bf16* __restrict__ W, int ldw,
    const float* __restrict__ bias,
    float* __restrict__ C, long long ldc,
    int M, int N, int K)
{
    const int lane  = threadIdx.x;              // 0..31
    const int wv    = threadIdx.y;              // 0..3
    const int tileN = (blockIdx.x * 4 + wv) * 32;
    const int tileM = blockIdx.y * 16;
    if (tileN >= N || tileM >= M) return;       // wave-uniform guard (EXEC all-1)

    const int hk  = lane >> 4;                  // 0/1 -> which K sub-block
    const int l15 = lane & 15;
    const int mrow = tileM + l15;
    const int arow = mrow < M ? mrow : M - 1;
    const int brow0 = (tileN + l15)      < N ? (tileN + l15)      : N - 1;
    const int brow1 = (tileN + 16 + l15) < N ? (tileN + 16 + l15) : N - 1;

    const float*  pa  = A + (size_t)arow  * lda + hk * 8;
    const __bf16* pw0 = W + (size_t)brow0 * ldw + hk * 8;
    const __bf16* pw1 = W + (size_t)brow1 * ldw + hk * 8;

    v8f acc0 = {};
    v8f acc1 = {};
    for (int k0 = 0; k0 < K; k0 += 32) {
        if (k0 + 32 < K) {                      // near-cache prefetch of next chunk
            __builtin_prefetch(pa  + k0 + 32, 0, 3);
            __builtin_prefetch(pw0 + k0 + 32, 0, 3);
            __builtin_prefetch(pw1 + k0 + 32, 0, 3);
        }
        // A fragment: load 16 f32, convert + pack to bf16 (shared by both WMMAs)
        float4 x0 = *(const float4*)(pa + k0);
        float4 x1 = *(const float4*)(pa + k0 + 4);
        float4 x2 = *(const float4*)(pa + k0 + 16);
        float4 x3 = *(const float4*)(pa + k0 + 20);
        Frag16 a;
        a.u[0] = pk2bf(x0.x, x0.y); a.u[1] = pk2bf(x0.z, x0.w);
        a.u[2] = pk2bf(x1.x, x1.y); a.u[3] = pk2bf(x1.z, x1.w);
        a.u[4] = pk2bf(x2.x, x2.y); a.u[5] = pk2bf(x2.z, x2.w);
        a.u[6] = pk2bf(x3.x, x3.y); a.u[7] = pk2bf(x3.z, x3.w);
        // B fragments: weights already bf16, two 16B loads each
        uint4 w0 = *(const uint4*)(pw0 + k0);
        uint4 w1 = *(const uint4*)(pw0 + k0 + 16);
        Frag16 b0;
        b0.u[0] = w0.x; b0.u[1] = w0.y; b0.u[2] = w0.z; b0.u[3] = w0.w;
        b0.u[4] = w1.x; b0.u[5] = w1.y; b0.u[6] = w1.z; b0.u[7] = w1.w;
        uint4 w2 = *(const uint4*)(pw1 + k0);
        uint4 w3 = *(const uint4*)(pw1 + k0 + 16);
        Frag16 b1;
        b1.u[0] = w2.x; b1.u[1] = w2.y; b1.u[2] = w2.z; b1.u[3] = w2.w;
        b1.u[4] = w3.x; b1.u[5] = w3.y; b1.u[6] = w3.z; b1.u[7] = w3.w;

        acc0 = __builtin_amdgcn_wmma_f32_16x16x32_bf16(
            false, a.v, false, b0.v, (short)0, acc0, false, false);
        acc1 = __builtin_amdgcn_wmma_f32_16x16x32_bf16(
            false, a.v, false, b1.v, (short)0, acc1, false, false);
    }

    // C/D layout: elem i -> row tileM + i + 8*hk, col tileN + (lane&15)
    const int ncol0 = tileN + l15;
    const int ncol1 = tileN + 16 + l15;
    const float bv0 = (bias && ncol0 < N) ? bias[ncol0] : 0.0f;
    const float bv1 = (bias && ncol1 < N) ? bias[ncol1] : 0.0f;
    #pragma unroll
    for (int i = 0; i < 8; ++i) {
        const int m = tileM + i + 8 * hk;
        if (m < M) {
            float* crow = C + (size_t)m * ldc;
            if (ncol0 < N) crow[ncol0] = acc0[i] + bv0;
            if (ncol1 < N) crow[ncol1] = acc1[i] + bv1;
        }
    }
}

// ---------------------------------------------------------------------------
// Setup kernels
// ---------------------------------------------------------------------------
__global__ void cvt_f32_to_bf16(const float* __restrict__ in,
                                __bf16* __restrict__ out, int n) {
    int i = blockIdx.x * blockDim.x + threadIdx.x;
    if (i < n) out[i] = (__bf16)in[i];
}

// Wg[n][k] = (k < E+F) ? W_ih[n][k] : W_hh[n][k-(E+F)]   (2048 x 3072, bf16)
__global__ void pack_wg(const float* __restrict__ Wih,
                        const float* __restrict__ Whh,
                        __bf16* __restrict__ Wg) {
    int idx = blockIdx.x * blockDim.x + threadIdx.x;
    if (idx >= 4 * kH * kX) return;
    int n = idx / kX, k = idx % kX;
    float v = (k < kE + kF) ? Wih[(size_t)n * (kE + kF) + k]
                            : Whh[(size_t)n * kH + (k - kE - kF)];
    Wg[idx] = (__bf16)v;
}

__global__ void add_vec(const float* __restrict__ a,
                        const float* __restrict__ b,
                        float* __restrict__ o, int n) {
    int i = blockIdx.x * blockDim.x + threadIdx.x;
    if (i < n) o[i] = a[i] + b[i];
}

// mean over regions: mf[b][f] = (1/R) sum_r img[b][r][f]
__global__ void mean_feats_k(const float* __restrict__ img,
                             float* __restrict__ mf) {
    int idx = blockIdx.x * blockDim.x + threadIdx.x;
    if (idx >= kB * kF) return;
    int b = idx / kF, f = idx % kF;
    const float* p = img + (size_t)b * kR * kF + f;
    float s = 0.0f;
    for (int r = 0; r < kR; ++r) s += p[(size_t)r * kF];
    mf[idx] = s * (1.0f / (float)kR);
}

// ---------------------------------------------------------------------------
// Per-timestep kernels
// ---------------------------------------------------------------------------
// scores[b][r] = sum_a tanh(feat_proj[b][r][a] + hid_proj[b][a]) * w_score[a] + b_score
// one wave32 per (b,r) pair
__global__ __launch_bounds__(256) void attn_scores_k(
    const float* __restrict__ fp, const float* __restrict__ hp,
    const float* __restrict__ wsc, const float* __restrict__ bsc,
    float* __restrict__ scores) {
    int wid  = (blockIdx.x * blockDim.x + threadIdx.x) >> 5;
    int lane = threadIdx.x & 31;
    if (wid >= kB * kR) return;
    int b = wid / kR, r = wid % kR;
    const float* f = fp + ((size_t)b * kR + r) * kA;
    const float* h = hp + (size_t)b * kA;
    float s = 0.0f;
    for (int a = lane; a < kA; a += 32)
        s += tanhf(f[a] + h[a]) * wsc[a];
    #pragma unroll
    for (int off = 16; off > 0; off >>= 1) s += __shfl_xor(s, off, 32);
    if (lane == 0) scores[wid] = s + bsc[0];
}

// per-b softmax over R, write alpha to d_out, then context[b][f] = sum_r alpha*img
__global__ __launch_bounds__(256) void softmax_context_k(
    const float* __restrict__ scores, const float* __restrict__ img,
    float* __restrict__ context, float* __restrict__ alpha_out, int t) {
    __shared__ float sm[256];
    __shared__ float s_alpha[kR];
    const int b = blockIdx.x, tid = threadIdx.x;
    float v = (tid < kR) ? scores[b * kR + tid] : -3.0e38f;
    sm[tid] = v; __syncthreads();
    for (int s = 128; s > 0; s >>= 1) {
        if (tid < s) sm[tid] = fmaxf(sm[tid], sm[tid + s]);
        __syncthreads();
    }
    const float mx = sm[0]; __syncthreads();
    float e = (tid < kR) ? __expf(v - mx) : 0.0f;
    sm[tid] = e; __syncthreads();
    for (int s = 128; s > 0; s >>= 1) {
        if (tid < s) sm[tid] += sm[tid + s];
        __syncthreads();
    }
    const float inv = 1.0f / sm[0];
    if (tid < kR) {
        float al = e * inv;
        s_alpha[tid] = al;
        alpha_out[((size_t)b * kL + t) * kR + tid] = al;
    }
    __syncthreads();
    const float* ib = img + (size_t)b * kR * kF;
    for (int f = tid; f < kF; f += 256) {
        float acc = 0.0f;
        for (int r = 0; r < kR; ++r) acc += s_alpha[r] * ib[(size_t)r * kF + f];
        context[(size_t)b * kF + f] = acc;
    }
}

// xcat[b] = [ emb_table[cap[b][t]] (E) | context[b] (F) | h[b] (H) ]
__global__ void build_xcat_k(const float* __restrict__ embT,
                             const int* __restrict__ cap,
                             const float* __restrict__ context,
                             const float* __restrict__ h,
                             int t, float* __restrict__ xcat) {
    int idx = blockIdx.x * blockDim.x + threadIdx.x;
    if (idx >= kB * kX) return;
    int b = idx / kX, j = idx % kX;
    float v;
    if (j < kE)            v = embT[(size_t)cap[b * kL + t] * kE + j];
    else if (j < kE + kF)  v = context[(size_t)b * kF + (j - kE)];
    else                   v = h[(size_t)b * kH + (j - kE - kF)];
    xcat[idx] = v;
}

// LSTM pointwise, gate order i,f,g,o ; updates c,h in place
__global__ void lstm_cell_k(const float* __restrict__ gates,
                            float* __restrict__ c, float* __restrict__ h) {
    int idx = blockIdx.x * blockDim.x + threadIdx.x;
    if (idx >= kB * kH) return;
    int b = idx / kH, j = idx % kH;
    const float* g = gates + (size_t)b * 4 * kH;
    const float ig = sigmoidf_(g[j]);
    const float fg = sigmoidf_(g[kH + j]);
    const float gg = tanhf(g[2 * kH + j]);
    const float og = sigmoidf_(g[3 * kH + j]);
    const float cn = fg * c[idx] + ig * gg;
    c[idx] = cn;
    h[idx] = og * tanhf(cn);
}

// ---------------------------------------------------------------------------
// Host
// ---------------------------------------------------------------------------
static inline void launch_gemm(const float* A, int lda, const __bf16* W, int ldw,
                               const float* bias, float* C, long long ldc,
                               int M, int N, int K, hipStream_t s) {
    dim3 blk(32, 4), grd((N + 127) / 128, (M + 15) / 16);
    gemm_bf16_wmma<<<grd, blk, 0, s>>>(A, lda, W, ldw, bias, C, ldc, M, N, K);
}

extern "C" void kernel_launch(void* const* d_in, const int* in_sizes, int n_in,
                              void* d_out, int out_size, void* d_ws, size_t ws_size,
                              hipStream_t stream) {
    const float* img     = (const float*)d_in[0];   // (B,R,F)
    const int*   cap     = (const int*)  d_in[1];   // (B,L)
    const float* embT    = (const float*)d_in[2];   // (V,E)
    const float* Winit_h = (const float*)d_in[3];
    const float* binit_h = (const float*)d_in[4];
    const float* Winit_c = (const float*)d_in[5];
    const float* binit_c = (const float*)d_in[6];
    const float* Wih     = (const float*)d_in[7];   // (4H, E+F)
    const float* bih     = (const float*)d_in[8];
    const float* Whh     = (const float*)d_in[9];   // (4H, H)
    const float* bhh     = (const float*)d_in[10];
    const float* Wfeatp  = (const float*)d_in[11];  // (A, F)
    const float* bfeatp  = (const float*)d_in[12];
    const float* Whidp   = (const float*)d_in[13];  // (A, H)
    const float* bhidp   = (const float*)d_in[14];
    const float* Wscore  = (const float*)d_in[15];  // (1, A)
    const float* bscore  = (const float*)d_in[16];
    const float* Wout    = (const float*)d_in[17];  // (V, H)
    const float* bout    = (const float*)d_in[18];

    float* out = (float*)d_out;
    float* alpha_out = out + (size_t)kB * kL * kV;  // alphas after predictions

    // workspace carve-up (everything read is rewritten every call)
    char* ws = (char*)d_ws;
    size_t off = 0;
    auto carve = [&](size_t bytes) -> char* {
        char* p = ws + off;
        off = (off + bytes + 255) & ~(size_t)255;
        return p;
    };
    float*  mf      = (float*) carve((size_t)kB * kF * 4);
    float*  fproj   = (float*) carve((size_t)kB * kR * kA * 4);
    float*  h       = (float*) carve((size_t)kB * kH * 4);
    float*  c       = (float*) carve((size_t)kB * kH * 4);
    float*  hidp    = (float*) carve((size_t)kB * kA * 4);
    float*  scores  = (float*) carve((size_t)kB * kR * 4);
    float*  context = (float*) carve((size_t)kB * kF * 4);
    float*  xcat    = (float*) carve((size_t)kB * kX * 4);
    float*  gates   = (float*) carve((size_t)kB * 4 * kH * 4);
    float*  biasg   = (float*) carve((size_t)4 * kH * 4);
    __bf16* Wfp_bf  = (__bf16*)carve((size_t)kA * kF * 2);
    __bf16* Wih_bf  = (__bf16*)carve((size_t)kH * kF * 2);   // W_init_h
    __bf16* Wic_bf  = (__bf16*)carve((size_t)kH * kF * 2);   // W_init_c
    __bf16* Whp_bf  = (__bf16*)carve((size_t)kA * kH * 2);   // W_hidp
    __bf16* Wo_bf   = (__bf16*)carve((size_t)kV * kH * 2);   // W_out
    __bf16* Wg_bf   = (__bf16*)carve((size_t)4 * kH * kX * 2);

    auto cvt = [&](const float* src, __bf16* dst, int n) {
        cvt_f32_to_bf16<<<(n + 255) / 256, 256, 0, stream>>>(src, dst, n);
    };

    // ---- one-time (per call) weight prep ----
    cvt(Wfeatp,  Wfp_bf, kA * kF);
    cvt(Winit_h, Wih_bf, kH * kF);
    cvt(Winit_c, Wic_bf, kH * kF);
    cvt(Whidp,   Whp_bf, kA * kH);
    cvt(Wout,    Wo_bf,  kV * kH);
    pack_wg<<<(4 * kH * kX + 255) / 256, 256, 0, stream>>>(Wih, Whh, Wg_bf);
    add_vec<<<(4 * kH + 255) / 256, 256, 0, stream>>>(bih, bhh, biasg, 4 * kH);
    mean_feats_k<<<(kB * kF + 255) / 256, 256, 0, stream>>>(img, mf);

    // h0 / c0 / feat_proj
    launch_gemm(mf,  kF, Wih_bf, kF, binit_h, h,     kH, kB,      kH, kF, stream);
    launch_gemm(mf,  kF, Wic_bf, kF, binit_c, c,     kH, kB,      kH, kF, stream);
    launch_gemm(img, kF, Wfp_bf, kF, bfeatp,  fproj, kA, kB * kR, kA, kF, stream);

    // ---- recurrence ----
    for (int t = 0; t < kL; ++t) {
        launch_gemm(h, kH, Whp_bf, kH, bhidp, hidp, kA, kB, kA, kH, stream);
        attn_scores_k<<<(kB * kR * 32 + 255) / 256, 256, 0, stream>>>(
            fproj, hidp, Wscore, bscore, scores);
        softmax_context_k<<<kB, 256, 0, stream>>>(scores, img, context, alpha_out, t);
        build_xcat_k<<<(kB * kX + 255) / 256, 256, 0, stream>>>(
            embT, cap, context, h, t, xcat);
        launch_gemm(xcat, kX, Wg_bf, kX, biasg, gates, 4 * kH, kB, 4 * kH, kX, stream);
        lstm_cell_k<<<(kB * kH + 255) / 256, 256, 0, stream>>>(gates, c, h);
        launch_gemm(h, kH, Wo_bf, kH, bout, out + (size_t)t * kV,
                    (long long)kL * kV, kB, kV, kH, stream);
    }
}
